// SinkhornLoss_82033875353948
// MI455X (gfx1250) — matrix operand order repaired and checked by reading, not compiled, not run on hardware
//
#include <hip/hip_runtime.h>

#define NN 8192
#define INV_EPS 10.0f
#define SINK_ITER 500

typedef __bf16 bf16_t;
typedef __bf16 bfx8  __attribute__((ext_vector_type(8)));
typedef __bf16 v16bf __attribute__((ext_vector_type(16)));
typedef float  v8f   __attribute__((ext_vector_type(8)));
typedef unsigned short usx8 __attribute__((ext_vector_type(8)));
typedef float  f4    __attribute__((ext_vector_type(4)));

// ---------------------------------------------------------------------------
// Kb[i,j] = bf16(exp(-M[i,j]/eps)).  128 MB -> resident in 192 MB L2.
// M is streamed non-temporally so it doesn't pollute L2.
// ---------------------------------------------------------------------------
__global__ void __launch_bounds__(256)
k_gibbs(const float* __restrict__ M, bf16_t* __restrict__ Kb) {
  size_t base = ((size_t)blockIdx.x * 256 + threadIdx.x) * 8;
  f4 m0 = __builtin_nontemporal_load((const f4*)(M + base));
  f4 m1 = __builtin_nontemporal_load((const f4*)(M + base + 4));
  bfx8 o;
#pragma unroll
  for (int e = 0; e < 4; ++e) {
    o[e]     = (bf16_t)__expf(-INV_EPS * m0[e]);
    o[4 + e] = (bf16_t)__expf(-INV_EPS * m1[e]);
  }
  *(bfx8*)(Kb + base) = o;
}

// ---------------------------------------------------------------------------
// sums[0]=sum(a), sums[1]=sum(b); zero the output accumulator.
// ---------------------------------------------------------------------------
__global__ void __launch_bounds__(512)
k_prep(const float* __restrict__ a, const float* __restrict__ b,
       float* __restrict__ sums, float* __restrict__ out) {
  __shared__ float sm[512];
  int tid = threadIdx.x;
  const float* src = (tid < 256) ? a : b;
  int t = tid & 255;
  float s = 0.f;
  for (int k = t; k < NN; k += 256) s += src[k];
  sm[tid] = s;
  __syncthreads();
  for (int st = 128; st > 0; st >>= 1) {
    if ((tid & 255) < st) sm[tid] += sm[tid + st];
    __syncthreads();
  }
  if (tid == 0) { sums[0] = sm[0]; sums[1] = sm[256]; out[0] = 0.f; }
}

__global__ void __launch_bounds__(256)
k_vinit(float* __restrict__ vf, bf16_t* __restrict__ vb) {
  int i = blockIdx.x * 256 + threadIdx.x;
  vf[i] = 1.0f;
  vb[i] = (bf16_t)1.0f;
}

// ---------------------------------------------------------------------------
// Phase A:  u = (a/sum_a) / (K @ v)
// Block = 4 waves, one 16-row strip of K; waves split the column range
// (j interleaved by 32*4).  Literal trip count (64) keeps the loop scalar
// (no divergent exec-mask loop), and unroll-by-4 with two accumulators lets
// the next tiles' global_load_b128s issue before the current WMMA's
// s_wait_loadcnt, overlapping L2 latency with the matrix pipe.
// ---------------------------------------------------------------------------
__global__ void __launch_bounds__(128)
k_phaseA(const bf16_t* __restrict__ Kb, const bf16_t* __restrict__ vb,
         const float* __restrict__ a, const float* __restrict__ sums,
         float* __restrict__ uf, bf16_t* __restrict__ ub) {
  __shared__ float ldsy[4][16];
  const int tid  = threadIdx.x;
  const int wv   = tid >> 5;
  const int lane = tid & 31;
  const int hi   = lane >> 4;
  const int n    = lane & 15;
  const int r0   = blockIdx.x * 16;
  // Per-lane base pointers; all further offsets are literal immediates.
  const bf16_t* pA = Kb + (size_t)(r0 + n) * NN + wv * 32 + 8 * hi;
  const bf16_t* pB = vb + wv * 32 + 8 * hi;

  v8f acc0 = {};
  v8f acc1 = {};
#pragma unroll 4
  for (int t = 0; t < 64; ++t) {
    const int o = t * 128; // 64 tiles, stride 4 waves * 32 cols
    union { v16bf v; bfx8 h[2]; } A, B;
    A.h[0] = *(const bfx8*)(pA + o);
    A.h[1] = *(const bfx8*)(pA + o + 16);
    B.h[0] = *(const bfx8*)(pB + o);
    B.h[1] = *(const bfx8*)(pB + o + 16);
    if (t & 1)
      acc1 = __builtin_amdgcn_wmma_f32_16x16x32_bf16(
          false, A.v, false, B.v, (short)0, acc1, false, false);
    else
      acc0 = __builtin_amdgcn_wmma_f32_16x16x32_bf16(
          false, A.v, false, B.v, (short)0, acc0, false, false);
  }
  v8f acc = acc0 + acc1;
  // D layout: lane group hi, VGPR r -> row M = r + 8*hi; all N columns equal.
  if (n == 0) {
#pragma unroll
    for (int r = 0; r < 8; ++r) ldsy[wv][r + 8 * hi] = acc[r];
  }
  __syncthreads();
  if (tid < 16) {
    float y = ldsy[0][tid] + ldsy[1][tid] + ldsy[2][tid] + ldsy[3][tid];
    int ri = r0 + tid;
    float uu = (a[ri] / sums[0]) / y;
    uf[ri] = uu;
    ub[ri] = (bf16_t)uu;
  }
}

// ---------------------------------------------------------------------------
// Phase B:  v = (b/sum_b) / (K^T @ u)
// Block = 8 waves, owns 64 columns.  Two sets of 4 waves split the row range
// (even/odd 32-row tiles).  Each set stages its 32x64 bf16 tile with fully
// coalesced 128B row reads, packs row-pairs into uint32, and writes a
// transposed LDS layout ldsT[col][rowpair] with stride 17 (conflict-free for
// both the packed stores and the 8 ds_load_b32 B-fragment reads).
// A fragment = u broadcast across the 16 rows.  Double-buffered staging.
// ---------------------------------------------------------------------------
__global__ void __launch_bounds__(256)
k_phaseB(const bf16_t* __restrict__ Kb, const bf16_t* __restrict__ ub,
         const float* __restrict__ b, const float* __restrict__ sums,
         float* __restrict__ vf, bf16_t* __restrict__ vb) {
  __shared__ unsigned int ldsT[2][2][64 * 17];
  __shared__ float ldsz[4][16];
  const int tid   = threadIdx.x;
  const int wv    = tid >> 5;
  const int lane  = tid & 31;
  const int set   = wv >> 2;   // 0: even 32-row tiles, 1: odd
  const int strip = wv & 3;    // which 16-column strip of the 64
  const int hi    = lane >> 4;
  const int n     = lane & 15;
  const int j0    = blockIdx.x * 64;
  const int t128  = tid & 127;
  const int rp    = t128 >> 3; // row-pair 0..15 within 32-row tile
  const int cc    = t128 & 7;  // 8-col chunk 0..7 within 64 cols

  auto stage = [&](int buf, int i) {
    const bf16_t* p = Kb + (size_t)(i + 2 * rp) * NN + j0 + 8 * cc;
    usx8 ra = *(const usx8*)p;        // row 2*rp
    usx8 rb = *(const usx8*)(p + NN); // row 2*rp+1
    unsigned int* dst = &ldsT[set][buf][0];
#pragma unroll
    for (int e = 0; e < 8; ++e)
      dst[(8 * cc + e) * 17 + rp] =
          (unsigned int)ra[e] | ((unsigned int)rb[e] << 16);
  };

  const int c = strip * 16 + n; // LDS column for this lane's B fragment
  v8f acc = {};
  stage(0, set * 32);
  int buf = 0;
  for (int t = 0; t < 128; ++t) {
    __syncthreads();
    if (t + 1 < 128) stage(buf ^ 1, (2 * (t + 1) + set) * 32);
    const int i = (2 * t + set) * 32;
    union { v16bf v; bfx8 h[2]; } A;
    union { v16bf v; unsigned int u[8]; } B;
    A.h[0] = *(const bfx8*)(ub + i + 16 * hi);
    A.h[1] = *(const bfx8*)(ub + i + 8 + 16 * hi);
#pragma unroll
    for (int r = 0; r < 8; ++r)
      B.u[r] = ldsT[set][buf][c * 17 + r + 8 * hi]; // k pair (2r,2r+1)+16*hi
    acc = __builtin_amdgcn_wmma_f32_16x16x32_bf16(
        false, A.v, false, B.v, (short)0, acc, false, false);
    buf ^= 1;
  }
  __syncthreads();
  // All D rows equal (A rows are identical broadcasts of u); z lives at acc[0].
  if (set == 1 && hi == 0) ldsz[strip][n] = acc[0];
  __syncthreads();
  if (set == 0 && hi == 0) {
    float z = acc[0] + ldsz[strip][n];
    int col = j0 + strip * 16 + n;
    float vv = (b[col] / sums[1]) / z;
    vf[col] = vv;
    vb[col] = (bf16_t)vv;
  }
}

// ---------------------------------------------------------------------------
// loss = sum_ij u_i * exp(-M/eps) * v_j * M_ij   (K recomputed in f32)
// ---------------------------------------------------------------------------
__global__ void __launch_bounds__(256)
k_loss(const float* __restrict__ M, const float* __restrict__ uf,
       const float* __restrict__ vf, float* __restrict__ out) {
  __shared__ float sm[256];
  size_t base = ((size_t)blockIdx.x * 256 + threadIdx.x) * 4;
  int row = (int)(base >> 13);   // /8192
  int col = (int)(base & 8191);
  f4 m = __builtin_nontemporal_load((const f4*)(M + base));
  float uo = uf[row];
  float s = 0.f;
#pragma unroll
  for (int e = 0; e < 4; ++e) {
    float mm = m[e];
    s += uo * vf[col + e] * __expf(-INV_EPS * mm) * mm;
  }
  sm[threadIdx.x] = s;
  __syncthreads();
  for (int st = 128; st > 0; st >>= 1) {
    if (threadIdx.x < st) sm[threadIdx.x] += sm[threadIdx.x + st];
    __syncthreads();
  }
  if (threadIdx.x == 0) atomicAdd(out, sm[0]);
}

// ---------------------------------------------------------------------------
extern "C" void kernel_launch(void* const* d_in, const int* in_sizes, int n_in,
                              void* d_out, int out_size, void* d_ws,
                              size_t ws_size, hipStream_t stream) {
  const float* a = (const float*)d_in[0];
  const float* b = (const float*)d_in[1];
  const float* M = (const float*)d_in[2];
  float* out = (float*)d_out;

  char* ws = (char*)d_ws;
  bf16_t* Kb = (bf16_t*)ws;                         // 128 MB bf16 Gibbs kernel
  size_t off = (size_t)NN * NN * sizeof(bf16_t);
  float*  uf = (float*)(ws + off);  off += (size_t)NN * sizeof(float);
  float*  vf = (float*)(ws + off);  off += (size_t)NN * sizeof(float);
  bf16_t* ub = (bf16_t*)(ws + off); off += (size_t)NN * sizeof(bf16_t);
  bf16_t* vb = (bf16_t*)(ws + off); off += (size_t)NN * sizeof(bf16_t);
  float* sums = (float*)(ws + off);

  k_gibbs<<<(NN * NN) / (256 * 8), 256, 0, stream>>>(M, Kb);
  k_prep<<<1, 512, 0, stream>>>(a, b, sums, out);
  k_vinit<<<NN / 256, 256, 0, stream>>>(vf, vb);
  for (int it = 0; it < SINK_ITER; ++it) {
    k_phaseA<<<NN / 16, 128, 0, stream>>>(Kb, vb, a, sums, uf, ub);
    k_phaseB<<<NN / 64, 256, 0, stream>>>(Kb, ub, b, sums, vf, vb);
  }
  k_loss<<<(NN * NN) / (256 * 4), 256, 0, stream>>>(M, uf, vf, out);
}
